// THAN_13692355740191
// MI455X (gfx1250) — matrix operand order repaired and checked by reading, not compiled, not run on hardware
//
#include <hip/hip_runtime.h>
#include <hip/hip_bf16.h>

// ---------------------------------------------------------------------------
// THAN (temporal graph attention) forward for MI455X / gfx1250.
// Heavy GEMMs run on v_wmma_f32_16x16x32_f16; attention/softmax/LN on VALU.
// ---------------------------------------------------------------------------

typedef _Float16 h16;
typedef __attribute__((ext_vector_type(16))) _Float16 v16h;
typedef __attribute__((ext_vector_type(8)))  _Float16 v8h;
typedef __attribute__((ext_vector_type(8)))  float    v8f;

__device__ __forceinline__ v8f wmma32(v16h a, v16h b, v8f c) {
    return __builtin_amdgcn_wmma_f32_16x16x32_f16(false, a, false, b, (short)0, c, false, false);
}

// A fragment (16x32 f16) from row-major source, ld in halfs.
// Contiguous k order used here (packed B matches), not the ISA table order: lane m=lane&15, kh=lane>>4; VGPR0-3: K = kh*16 + 0..7, VGPR4-7: K = kh*16 + 8 + 0..7.
__device__ __forceinline__ v16h load_a16(const h16* tile, int ld, int lane) {
    int m = lane & 15, kh = lane >> 4;
    const v8h* p0 = (const v8h*)(tile + (size_t)m * ld + kh * 16);
    const v8h* p1 = (const v8h*)(tile + (size_t)m * ld + kh * 16 + 8);
    v8h lo = *p0, hi = *p1;
    v16h r;
#pragma unroll
    for (int i = 0; i < 8; i++) { r[i] = lo[i]; r[8 + i] = hi[i]; }
    return r;
}

// B fragment from pre-packed layout: 16 consecutive halfs per lane per (ktile,ntile) block.
__device__ __forceinline__ v16h load_b16(const h16* pk, int bk, int bn, int Ntot, int lane) {
    const v16h* p = (const v16h*)(pk + ((size_t)(bk * (Ntot >> 4) + bn) * 32 + lane) * 16);
    return *p;
}

// Packed-B offset for element (k, n) of a KxNtot matrix.
__device__ __forceinline__ size_t pack_off(int k, int n, int Ntot) {
    return ((size_t)((k >> 5) * (Ntot >> 4) + (n >> 4)) * 32 + (((k >> 4) & 1) * 16 + (n & 15))) * 16 + (k & 15);
}

// ---------- prep kernels ----------------------------------------------------

// Pack a K x Nsrc f32 region (rows koff..koff+K of src, ld=ldsrc) into packed f16 dst
// at column offset noff inside a Ntot-wide packed matrix.
__global__ void pack_k(h16* dst, const float* src, int K, int Nsrc, int Ntot,
                       int noff, int koff, int ldsrc) {
    int i = blockIdx.x * blockDim.x + threadIdx.x;
    if (i >= K * Nsrc) return;
    int k = i / Nsrc, n = i % Nsrc;
    dst[pack_off(k, noff + n, Ntot)] = (h16)src[(size_t)(koff + k) * ldsrc + n];
}

// Combined typed weights: CKV[t] = We[t](128x128) @ [wk_top | wv_top] (128x768), packed f16.
__global__ void ckv_k(h16* dst, const float* We, const float* wk, const float* wv) {
    int i = blockIdx.x * blockDim.x + threadIdx.x;
    if (i >= 4 * 128 * 768) return;
    int t = i / 98304, rem = i % 98304, k = rem / 768, n = rem % 768;
    const float* w   = We + (size_t)t * 16384 + (size_t)k * 128;
    const float* col = (n < 384) ? (wk + n) : (wv + (n - 384));
    float s = 0.f;
    for (int m = 0; m < 128; m++) s += w[m] * col[(size_t)m * 384];
    dst[(size_t)t * 98304 + pack_off(k, n, 768)] = (h16)s;
}

// qbias[n] = sum_{k=256..383} wq[k][n]   (src_t == cos(0) == 1)
__global__ void qbias_k(float* qb, const float* wq) {
    int n = blockIdx.x * blockDim.x + threadIdx.x;
    if (n >= 384) return;
    float s = 0.f;
    for (int k = 256; k < 384; k++) s += wq[(size_t)k * 384 + n];
    qb[n] = s;
}

// ---------- typed src transform: src_h = n_feat[node] @ Wn[utype] -----------
__global__ __launch_bounds__(128) void srch_k(float* of, h16* oh,
        const float* n_feat, const float* Wn,
        const int* idxA, const int* idxB, int split,
        const int* utA, const int* utB) {
    int b = blockIdx.x, tid = threadIdx.x;
    __shared__ float xr[128];
    int node = (b < split) ? idxA[b] : idxB[b - split];
    int ut   = (b < split) ? utA[b]  : utB[b - split];
    xr[tid] = n_feat[(size_t)node * 128 + tid];
    __syncthreads();
    const float* W = Wn + (size_t)ut * 16384;
    float s = 0.f;
    for (int m = 0; m < 128; m++) s += xr[m] * W[(size_t)m * 128 + tid];
    of[(size_t)b * 128 + tid] = s;
    oh[(size_t)b * 128 + tid] = (h16)s;
}

// ---------- generic WMMA GEMM: out = A(f16, MxK) @ Bpk(packed f16, KxN) -----
// grid = (M/16, N/64), 32 threads. flags&1 = ReLU. bias nullable, outs nullable.
__global__ __launch_bounds__(32) void gemm16_k(float* outf, h16* outh,
        const h16* A, const h16* Bpk, const float* bias,
        int N, int K, int flags) {
    int lane = threadIdx.x;
    int mb = blockIdx.x, nb = blockIdx.y;
    const h16* Arow = A + (size_t)mb * 16 * K;
    v8f acc[4] = {};
    for (int kb = 0; kb < K; kb += 32) {
        v16h a = load_a16(Arow + kb, K, lane);
#pragma unroll
        for (int t = 0; t < 4; t++)
            acc[t] = wmma32(a, load_b16(Bpk, kb >> 5, nb * 4 + t, N, lane), acc[t]);
    }
    int c = lane & 15, rb = mb * 16 + (lane >> 4) * 8;
#pragma unroll
    for (int t = 0; t < 4; t++) {
        int n = nb * 64 + t * 16 + c;
        float bb = bias ? bias[n] : 0.f;
#pragma unroll
        for (int v = 0; v < 8; v++) {
            float x = acc[t][v] + bb;
            if (flags & 1) x = fmaxf(x, 0.f);
            size_t o = (size_t)(rb + v) * N + n;
            if (outf) outf[o] = x;
            if (outh) outh[o] = (h16)x;
        }
    }
}

// ---------- fused K|V projection + attention per source node ----------------
// One workgroup (12 waves = 384 thr) per b. Stages x0/e_feat/time-enc rows in
// LDS (f16), runs the 768-wide masked-typed GEMM (288 WMMA/wave), writes K|V
// (48x768 f16) to LDS, then does 4-head attention over 40 neighbors and emits
// o[b, 384] in f16.
__global__ __launch_bounds__(384) void build_kv_k(
        h16* O16, const float* Qg,
        const float* x0src, const int* x0idx,
        const float* e_feat, const int* eidx,
        const float* freq, const float* tarr,
        const float* cut, int cut_mod,
        const int* etyp, const int* nodeid,
        const h16* CKV, const h16* EKV, const h16* TKV) {
    __shared__ alignas(16) h16 s_u[48 * 768];   // union: staging (18432 halfs) then K|V
    __shared__ float s_q[384];
    __shared__ float s_logit[4 * 48];
    __shared__ float s_a[4 * 48];
    __shared__ int   s_et[48];
    __shared__ float s_msk[48];

    h16* s_x0 = s_u;             // 48x128
    h16* s_ef = s_u + 6144;      // 48x128
    h16* s_st = s_u + 12288;     // 48x128
    h16* s_kv = s_u;             // 48x768 (after barrier)

    const int b   = blockIdx.x;
    const int tid = threadIdx.x;
    const int r0  = b * 40;
    const float cutb = cut[cut_mod ? (b % cut_mod) : b];

    for (int i = tid; i < 6144; i += 384) {
        int j = i >> 7, d = i & 127;
        float xv = 0.f, ev = 0.f, tv = 0.f;
        if (j < 40) {
            int r = r0 + j;
            int xrow = x0idx ? x0idx[r] : r;
            xv = x0src[(size_t)xrow * 128 + d];
            ev = e_feat[(size_t)eidx[r] * 128 + d];
            tv = cosf((cutb - tarr[r]) * freq[d]);
        }
        s_x0[i] = (h16)xv; s_ef[i] = (h16)ev; s_st[i] = (h16)tv;
    }
    if (tid < 48) {
        s_et[tid]  = (tid < 40) ? etyp[r0 + tid] : -1;
        s_msk[tid] = (tid < 40 && nodeid[r0 + tid] == 0) ? -1e10f : 0.f;
    }
    s_q[tid] = Qg[(size_t)b * 384 + tid];
    __syncthreads();

    // ---- GEMM: wave wid handles m-tile (wid%3), n-block (wid/3, 12 tiles) ----
    const int lane = tid & 31, wid = tid >> 5;
    const int mt = wid % 3, nq = wid / 3;
    v8f acc[12] = {};

    // phase 1: typed x0 @ CKV[t] with per-row type mask on A
#pragma unroll
    for (int t = 0; t < 4; t++) {
        bool keep = (s_et[mt * 16 + (lane & 15)] == t);
        const h16* bp = CKV + (size_t)t * 98304;
#pragma unroll
        for (int ks = 0; ks < 4; ks++) {
            v16h a = load_a16(s_x0 + (size_t)mt * 16 * 128 + ks * 32, 128, lane);
            if (!keep) {
#pragma unroll
                for (int i = 0; i < 16; i++) a[i] = (h16)0;
            }
#pragma unroll
            for (int nt = 0; nt < 12; nt++)
                acc[nt] = wmma32(a, load_b16(bp, ks, nq * 12 + nt, 768, lane), acc[nt]);
        }
    }
    // phase 2: e_feat @ EKV
#pragma unroll
    for (int ks = 0; ks < 4; ks++) {
        v16h a = load_a16(s_ef + (size_t)mt * 16 * 128 + ks * 32, 128, lane);
#pragma unroll
        for (int nt = 0; nt < 12; nt++)
            acc[nt] = wmma32(a, load_b16(EKV, ks, nq * 12 + nt, 768, lane), acc[nt]);
    }
    // phase 3: time enc @ TKV
#pragma unroll
    for (int ks = 0; ks < 4; ks++) {
        v16h a = load_a16(s_st + (size_t)mt * 16 * 128 + ks * 32, 128, lane);
#pragma unroll
        for (int nt = 0; nt < 12; nt++)
            acc[nt] = wmma32(a, load_b16(TKV, ks, nq * 12 + nt, 768, lane), acc[nt]);
    }
    __syncthreads();   // done reading staging; s_u becomes K|V

    // C layout: col = lane&15, rows = (lane>>4)*8 + v
#pragma unroll
    for (int nt = 0; nt < 12; nt++) {
        int n  = nq * 192 + nt * 16 + (lane & 15);
        int rb = mt * 16 + (lane >> 4) * 8;
#pragma unroll
        for (int v = 0; v < 8; v++) s_kv[(size_t)(rb + v) * 768 + n] = (h16)acc[nt][v];
    }
    __syncthreads();

    // ---- attention: logits, softmax over 40, o = a @ V ----
    if (tid < 160) {
        int h = tid / 40, j = tid % 40;
        const h16* kr = s_kv + (size_t)j * 768 + h * 96;
        float s = 0.f;
        for (int k = 0; k < 96; k++) s += s_q[h * 96 + k] * (float)kr[k];
        s_logit[h * 48 + j] = s * 0.1020620726f + s_msk[j];   // 1/sqrt(96)
    }
    __syncthreads();
    if (tid < 4) {
        int h = tid;
        float mx = -1e30f;
        for (int j = 0; j < 40; j++) mx = fmaxf(mx, s_logit[h * 48 + j]);
        float sm = 0.f;
        for (int j = 0; j < 40; j++) { float e = expf(s_logit[h * 48 + j] - mx); s_a[h * 48 + j] = e; sm += e; }
        float inv = 1.f / sm;
        for (int j = 0; j < 40; j++) s_a[h * 48 + j] *= inv;
    }
    __syncthreads();
    {
        int d = tid, h = d / 96;
        float o = 0.f;
        for (int j = 0; j < 40; j++) o += s_a[h * 48 + j] * (float)s_kv[(size_t)j * 768 + 384 + d];
        O16[(size_t)b * 384 + d] = (h16)o;
    }
}

// ---------- LayerNorm + build merge input [ln(o@fc+b+q) | src] --------------
__global__ __launch_bounds__(128) void ln_k(h16* X2, const float* FCo, const float* srch,
        const float* g, const float* beta, const float* n_feat,
        const int* idxA, const int* idxB, int split) {
    int b = blockIdx.x, tid = threadIdx.x;
    __shared__ float xr[384];
    __shared__ float red[128], red2[128];
    float s = 0.f, s2 = 0.f;
    for (int i = tid; i < 384; i += 128) {
        float v = FCo[(size_t)b * 384 + i];
        if (i < 128)       v += srch[(size_t)b * 128 + i];   // q residual: [src_h | 0 | 1]
        else if (i >= 256) v += 1.0f;
        xr[i] = v; s += v; s2 += v * v;
    }
    red[tid] = s; red2[tid] = s2;
    __syncthreads();
    for (int w = 64; w > 0; w >>= 1) {
        if (tid < w) { red[tid] += red[tid + w]; red2[tid] += red2[tid + w]; }
        __syncthreads();
    }
    float mu  = red[0] * (1.f / 384.f);
    float var = red2[0] * (1.f / 384.f) - mu * mu;
    float inv = rsqrtf(var + 1e-5f);
    for (int i = tid; i < 384; i += 128)
        X2[(size_t)b * 512 + i] = (h16)((xr[i] - mu) * inv * g[i] + beta[i]);
    int node = (b < split) ? idxA[b] : idxB[b - split];
    X2[(size_t)b * 512 + 384 + tid] = (h16)n_feat[(size_t)node * 128 + tid];
}

// ---------- decoder ---------------------------------------------------------
__global__ __launch_bounds__(128) void dec_k(float* out, const float* emb2,
        const float* eemb, const int* etyp,
        const float* w1, const float* b1, const float* w2, const float* b2) {
    int i = blockIdx.x, n = threadIdx.x;
    __shared__ float red[128];
    const float* s = emb2 + (size_t)i * 128;
    const float* t = emb2 + (size_t)(32 + i) * 128;
    const float* e = eemb + (size_t)etyp[i] * 128;
    float acc = b1[n];
    for (int k = 0; k < 128; k++) acc += s[k] * w1[(size_t)k * 128 + n];
    for (int k = 0; k < 128; k++) acc += t[k] * w1[(size_t)(128 + k) * 128 + n];
    for (int k = 0; k < 128; k++) acc += e[k] * w1[(size_t)(256 + k) * 128 + n];
    red[n] = fmaxf(acc, 0.f) * w2[n];
    __syncthreads();
    for (int w = 64; w > 0; w >>= 1) { if (n < w) red[n] += red[n + w]; __syncthreads(); }
    if (n == 0) out[i] = 1.f / (1.f + expf(-(red[0] + b2[0])));
}

// ---------------------------------------------------------------------------
extern "C" void kernel_launch(void* const* d_in, const int* in_sizes, int n_in,
                              void* d_out, int out_size, void* d_ws, size_t ws_size,
                              hipStream_t stream) {
    (void)in_sizes; (void)n_in; (void)out_size;
    // params tree-flattened (sorted dict keys): attn[0], attn[1], dec_*, etype_emb, time_freq, transfer_*
#define PF(i) ((const float*)d_in[(i)])
#define PI(i) ((const int*)d_in[(i)])
    const float* p_fc[2]   = {PF(0), PF(11)};
    const float* p_fcb[2]  = {PF(1), PF(12)};
    const float* p_lnb[2]  = {PF(2), PF(13)};
    const float* p_lng[2]  = {PF(3), PF(14)};
    const float* p_mgb1[2] = {PF(4), PF(15)};
    const float* p_mgb2[2] = {PF(5), PF(16)};
    const float* p_mgw1[2] = {PF(6), PF(17)};
    const float* p_mgw2[2] = {PF(7), PF(18)};
    const float* p_wk[2]   = {PF(8), PF(19)};
    const float* p_wq[2]   = {PF(9), PF(20)};
    const float* p_wv[2]   = {PF(10), PF(21)};
    const float* DEC_B1 = PF(22); const float* DEC_B2 = PF(23);
    const float* DEC_W1 = PF(24); const float* DEC_W2 = PF(25);
    const float* ETYPE_EMB = PF(26); const float* FREQ = PF(27);
    const float* TRE = PF(28); const float* TRN = PF(29);
    const float* NF = PF(30); const float* EF = PF(31);
    const int* SRC_IDX = PI(32); const int* TGT_IDX = PI(33);
    const float* CUT_TIME = PF(34);
    const int* ETYPE = PI(35); const int* SRC_UT = PI(36); const int* TGT_UT = PI(37);
    const int* NGH_NODE2 = PI(38); const int* NGH_EIDX2 = PI(39);
    const float* NGH_T2 = PF(40);
    const int* NGH_ETYPE2 = PI(41); const int* NGH_VTYPE2 = PI(42);
    const int* NGH_NODE1 = PI(43); const int* NGH_EIDX1 = PI(44);
    const float* NGH_T1 = PF(45);
    const int* NGH_ETYPE1 = PI(46);

    // ---- workspace bump allocator ----
    char* ws = (char*)d_ws;
    size_t off = 0;
    auto alloc = [&](size_t bytes) -> char* {
        char* p = ws + off; off += (bytes + 255) & ~(size_t)255; return p;
    };
    h16*  CKV  = (h16*)alloc(2 * 4 * 98304 * sizeof(h16));   // combined typed [wk|wv]_top
    h16*  EKV  = (h16*)alloc(2 * 98304 * sizeof(h16));
    h16*  TKV  = (h16*)alloc(2 * 98304 * sizeof(h16));
    h16*  WQp  = (h16*)alloc(2 * 49152 * sizeof(h16));
    h16*  FCp  = (h16*)alloc(2 * 147456 * sizeof(h16));
    h16*  M1p  = (h16*)alloc(2 * 65536 * sizeof(h16));
    h16*  M2p  = (h16*)alloc(2 * 16384 * sizeof(h16));
    float* QB  = (float*)alloc(2 * 384 * sizeof(float));
    float* SRH1f = (float*)alloc((size_t)2560 * 128 * 4);
    h16*   SRH1h = (h16*)alloc((size_t)2560 * 128 * 2);
    float* SRH2f = (float*)alloc((size_t)64 * 128 * 4);
    h16*   SRH2h = (h16*)alloc((size_t)64 * 128 * 2);
    float* Q1  = (float*)alloc((size_t)2560 * 384 * 4);
    float* Q2  = (float*)alloc((size_t)64 * 384 * 4);
    h16*   O1h = (h16*)alloc((size_t)2560 * 384 * 2);
    h16*   O2h = (h16*)alloc((size_t)64 * 384 * 2);
    float* FC1o = (float*)alloc((size_t)2560 * 384 * 4);
    float* FC2o = (float*)alloc((size_t)64 * 384 * 4);
    h16*   X21 = (h16*)alloc((size_t)2560 * 512 * 2);
    h16*   X22 = (h16*)alloc((size_t)64 * 512 * 2);
    h16*   H11 = (h16*)alloc((size_t)2560 * 128 * 2);
    h16*   H12 = (h16*)alloc((size_t)64 * 128 * 2);
    float* EMB1 = (float*)alloc((size_t)2560 * 128 * 4);
    float* EMB2 = (float*)alloc((size_t)64 * 128 * 4);
    if (off > ws_size) return;   // workspace too small; bail deterministically

    // ---- prep: pack weights / combined matrices (cheap, rerun every call) ----
    for (int l = 0; l < 2; l++) {
        h16* ekv = EKV + (size_t)l * 98304;
        h16* tkv = TKV + (size_t)l * 98304;
        pack_k<<<192, 256, 0, stream>>>(ekv, p_wk[l], 128, 384, 768, 0,   128, 384);
        pack_k<<<192, 256, 0, stream>>>(ekv, p_wv[l], 128, 384, 768, 384, 128, 384);
        pack_k<<<192, 256, 0, stream>>>(tkv, p_wk[l], 128, 384, 768, 0,   256, 384);
        pack_k<<<192, 256, 0, stream>>>(tkv, p_wv[l], 128, 384, 768, 384, 256, 384);
        pack_k<<<192, 256, 0, stream>>>(WQp + (size_t)l * 49152, p_wq[l],   128, 384, 384, 0, 0, 384);
        pack_k<<<576, 256, 0, stream>>>(FCp + (size_t)l * 147456, p_fc[l],  384, 384, 384, 0, 0, 384);
        pack_k<<<256, 256, 0, stream>>>(M1p + (size_t)l * 65536, p_mgw1[l], 512, 128, 128, 0, 0, 128);
        pack_k<<<64,  256, 0, stream>>>(M2p + (size_t)l * 16384, p_mgw2[l], 128, 128, 128, 0, 0, 128);
        ckv_k<<<1536, 256, 0, stream>>>(CKV + (size_t)l * 4 * 98304, TRE, p_wk[l], p_wv[l]);
        qbias_k<<<3, 128, 0, stream>>>(QB + l * 384, p_wq[l]);
    }

    const int BIG = 1 << 30;
    // ---- layer 1: B=2560 sources, 40 neighbors each ----
    srch_k<<<2560, 128, 0, stream>>>(SRH1f, SRH1h, NF, TRN, NGH_NODE2, nullptr, BIG,
                                     NGH_VTYPE2, nullptr);
    gemm16_k<<<dim3(160, 6), 32, 0, stream>>>(Q1, nullptr, SRH1h, WQp, QB, 384, 128, 0);
    build_kv_k<<<2560, 384, 0, stream>>>(O1h, Q1, NF, NGH_NODE1, EF, NGH_EIDX1, FREQ,
                                         NGH_T1, NGH_T2, 0, NGH_ETYPE1, NGH_NODE1,
                                         CKV, EKV, TKV);
    gemm16_k<<<dim3(160, 6), 32, 0, stream>>>(FC1o, nullptr, O1h, FCp, p_fcb[0], 384, 384, 0);
    ln_k<<<2560, 128, 0, stream>>>(X21, FC1o, SRH1f, p_lng[0], p_lnb[0], NF,
                                   NGH_NODE2, nullptr, BIG);
    gemm16_k<<<dim3(160, 2), 32, 0, stream>>>(nullptr, H11, X21, M1p, p_mgb1[0], 128, 512, 1);
    gemm16_k<<<dim3(160, 2), 32, 0, stream>>>(EMB1, nullptr, H11, M2p, p_mgb2[0], 128, 128, 0);

    // ---- layer 2: B=64 sources, seq = emb1 ----
    srch_k<<<64, 128, 0, stream>>>(SRH2f, SRH2h, NF, TRN, SRC_IDX, TGT_IDX, 32,
                                   SRC_UT, TGT_UT);
    gemm16_k<<<dim3(4, 6), 32, 0, stream>>>(Q2, nullptr, SRH2h, WQp + 49152, QB + 384, 384, 128, 0);
    build_kv_k<<<64, 384, 0, stream>>>(O2h, Q2, EMB1, nullptr, EF, NGH_EIDX2, FREQ,
                                       NGH_T2, CUT_TIME, 32, NGH_ETYPE2, NGH_NODE2,
                                       CKV + (size_t)4 * 98304, EKV + 98304, TKV + 98304);
    gemm16_k<<<dim3(4, 6), 32, 0, stream>>>(FC2o, nullptr, O2h, FCp + 147456, p_fcb[1], 384, 384, 0);
    ln_k<<<64, 128, 0, stream>>>(X22, FC2o, SRH2f, p_lng[1], p_lnb[1], NF,
                                 SRC_IDX, TGT_IDX, 32);
    gemm16_k<<<dim3(4, 2), 32, 0, stream>>>(nullptr, H12, X22, M1p + 65536, p_mgb1[1], 128, 512, 1);
    gemm16_k<<<dim3(4, 2), 32, 0, stream>>>(EMB2, nullptr, H12, M2p + 16384, p_mgb2[1], 128, 128, 0);

    // ---- decoder ----
    dec_k<<<32, 128, 0, stream>>>((float*)d_out, EMB2, ETYPE_EMB, ETYPE,
                                  DEC_W1, DEC_B1, DEC_W2, DEC_B2);
#undef PF
#undef PI
}
